// CNN_Transformer_60017872995050
// MI455X (gfx1250) — matrix operand order repaired and checked by reading, not compiled, not run on hardware
//
#include <hip/hip_runtime.h>
#include <math.h>

// ---------------------------------------------------------------------------
// CNN(VGG16)+Transformer forward for MI455X (gfx1250, wave32, WMMA).
// Heavy convs: implicit-GEMM with v_wmma_f32_16x16x32_f16, NHWC fp16
// activations, fp16 weights, f32 accumulation, 32x32 register tile per wave
// with software-pipelined (double-buffered) K loop. Tiny transformer in fp32.
// Requires ws_size >= ~760 MB (2x336MB ping/pong + fp16 weights + small bufs).
// ---------------------------------------------------------------------------

typedef __attribute__((ext_vector_type(16))) _Float16 v16h;
typedef __attribute__((ext_vector_type(8)))  _Float16 h8;
typedef __attribute__((ext_vector_type(8)))  float    v8f;

__device__ __forceinline__ v8f zero8() {
    v8f z;
#pragma unroll
    for (int i = 0; i < 8; ++i) z[i] = 0.0f;
    return z;
}
__device__ __forceinline__ v16h zero16() {
    v16h z;
#pragma unroll
    for (int i = 0; i < 16; ++i) z[i] = (_Float16)0.0f;
    return z;
}

// A-matrix (16x32 f16) fragment in contiguous k order (same as load_fragB):
// lane half=lane>>4, row m=lane&15.
// elems a[0..7]  = K[kk + half*16 .. +7]
// elems a[8..15] = K[kk + half*16 + 8 .. +7]
__device__ __forceinline__ v16h load_fragA(const _Float16* __restrict__ p, int half) {
    h8 lo = *(const h8*)(p + half * 16);
    h8 hi = *(const h8*)(p + half * 16 + 8);
    v16h f;
#pragma unroll
    for (int j = 0; j < 8; ++j) { f[j] = lo[j]; f[j + 8] = hi[j]; }
    return f;
}

// B-matrix (32x16 f16) fragment: lane = column n, contiguous 16 K values
// starting at kk + half*16 (same k order as load_fragA).
__device__ __forceinline__ v16h load_fragB(const _Float16* __restrict__ p, int half) {
    return *(const v16h*)(p + half * 16);
}

// conv B fragment with implicit im2col + SAME-padding: k -> (tap, ci) by
// shift/mask (Cin = 1<<cshift), per-lane pixel column xcol.
__device__ __forceinline__ v16h load_convB(const _Float16* __restrict__ inImg,
                                           int y, int xcol, int H, int W,
                                           int k, int cshift, int half)
{
    const int tap = k >> cshift;
    const int ci  = k & ((1 << cshift) - 1);
    const int r = tap / 3 - 1;
    const int s = tap - (tap / 3) * 3 - 1;
    const int yy = y + r, xx = xcol + s;
    if (yy >= 0 && yy < H && xx >= 0 && xx < W)
        return *(const v16h*)(inImg + ((size_t)yy * W + xx) * ((size_t)1 << cshift)
                              + ci + half * 16);
    return zero16();
}

// ---------------------------------------------------------------------------
// Generic TN GEMM:  out[n][m] = act( sum_k A[m][k]*B[n][k] + bias[m] )
// A fp16 [M][K] row-major, B fp16 [N][K] row-major, K%32==0, M%16==0.
// One wave per 16x16 tile. Writes fp16 and/or fp32 outputs (row-major [N][M]).
// ---------------------------------------------------------------------------
__global__ __launch_bounds__(32)
void gemm_tn_wmma(const _Float16* __restrict__ A, const _Float16* __restrict__ B,
                  const float* __restrict__ bias, _Float16* __restrict__ out16,
                  float* __restrict__ out32, int M, int N, int K, int do_relu)
{
    const int n0 = blockIdx.x << 4;
    const int m0 = blockIdx.y << 4;
    const int lane = threadIdx.x;
    const int half = lane >> 4;
    const int mn = lane & 15;

    int nload = n0 + mn;
    if (nload >= N) nload = N - 1;               // clamp loads; stores guarded
    const _Float16* arow = A + (size_t)(m0 + mn) * K;
    const _Float16* brow = B + (size_t)nload * K;

    v8f acc = zero8();
    for (int kk = 0; kk < K; kk += 32) {
        if (kk + 128 < K) __builtin_prefetch(arow + kk + 128, 0, 1);
        v16h a = load_fragA(arow + kk, half);
        v16h b = load_fragB(brow + kk, half);
        acc = __builtin_amdgcn_wmma_f32_16x16x32_f16(false, a, false, b,
                                                     (short)0, acc, false, false);
    }

    const int n = n0 + mn;
    if (n < N) {
        const int co = m0 + half * 8;            // D: VGPR g -> row co+g, lane -> col n
        if (out16) {
            h8 o;
#pragma unroll
            for (int g = 0; g < 8; ++g) {
                float v = acc[g] + (bias ? bias[co + g] : 0.0f);
                if (do_relu) v = fmaxf(v, 0.0f);
                o[g] = (_Float16)v;
            }
            *(h8*)(out16 + (size_t)n * M + co) = o;
        }
        if (out32) {
#pragma unroll
            for (int g = 0; g < 8; ++g) {
                float v = acc[g] + (bias ? bias[co + g] : 0.0f);
                if (do_relu) v = fmaxf(v, 0.0f);
                out32[(size_t)n * M + co + g] = v;
            }
        }
    }
}

// ---------------------------------------------------------------------------
// 3x3 SAME conv, implicit GEMM. in: NHWC fp16 [40][H][W][Cin], Cin=1<<cshift,
// Cin%32==0, Cout%32==0. w16: [Cout][9*Cin] fp16 with k=(r*3+s)*Cin+ci.
// One wave computes a 32-cout x 32-pixel tile (two 16-pixel subtiles) with
// 4 accumulators and a double-buffered K loop: next chunk's 8x b128 loads are
// issued before the current chunk's 4 WMMAs so loads overlap matrix ops.
// ---------------------------------------------------------------------------
__global__ __launch_bounds__(32)
void conv3x3_wmma(const _Float16* __restrict__ in, const _Float16* __restrict__ w16,
                  const float* __restrict__ bias, _Float16* __restrict__ out,
                  int H, int W, int cshift, int Cout)
{
    const int Cin = 1 << cshift;
    const int tilesX = W >> 4;
    const int pairsPerImg = (H * W) >> 5;
    int bx = blockIdx.x;
    const int pr  = bx % pairsPerImg;
    const int img = bx / pairsPerImg;
    const int t0 = pr * 2, t1 = t0 + 1;          // two 16-pixel subtiles
    const int y0 = t0 / tilesX, x0 = (t0 % tilesX) << 4;
    const int y1 = t1 / tilesX, x1 = (t1 % tilesX) << 4;
    const int co0 = blockIdx.y << 5;

    const int lane = threadIdx.x;
    const int half = lane >> 4;
    const int mn = lane & 15;
    const int K = 9 << cshift;

    const _Float16* wr0 = w16 + (size_t)(co0 + mn) * K;
    const _Float16* wr1 = w16 + (size_t)(co0 + 16 + mn) * K;
    const _Float16* inImg = in + (size_t)img * H * W * Cin;
    const int xc0 = x0 + mn, xc1 = x1 + mn;

    v8f acc00 = zero8(), acc01 = zero8(), acc10 = zero8(), acc11 = zero8();

    // prologue: chunk k=0
    v16h a0 = load_fragA(wr0, half);
    v16h a1 = load_fragA(wr1, half);
    v16h b0 = load_convB(inImg, y0, xc0, H, W, 0, cshift, half);
    v16h b1 = load_convB(inImg, y1, xc1, H, W, 0, cshift, half);

    int k = 0;
    while (k + 32 < K) {
        const int kn = k + 32;
        // preload next chunk (issues before this chunk's WMMAs -> overlap)
        v16h na0 = load_fragA(wr0 + kn, half);
        v16h na1 = load_fragA(wr1 + kn, half);
        v16h nb0 = load_convB(inImg, y0, xc0, H, W, kn, cshift, half);
        v16h nb1 = load_convB(inImg, y1, xc1, H, W, kn, cshift, half);

        acc00 = __builtin_amdgcn_wmma_f32_16x16x32_f16(false, a0, false, b0,
                                                       (short)0, acc00, false, false);
        acc10 = __builtin_amdgcn_wmma_f32_16x16x32_f16(false, a1, false, b0,
                                                       (short)0, acc10, false, false);
        acc01 = __builtin_amdgcn_wmma_f32_16x16x32_f16(false, a0, false, b1,
                                                       (short)0, acc01, false, false);
        acc11 = __builtin_amdgcn_wmma_f32_16x16x32_f16(false, a1, false, b1,
                                                       (short)0, acc11, false, false);
        a0 = na0; a1 = na1; b0 = nb0; b1 = nb1;
        k = kn;
    }
    // epilogue chunk
    acc00 = __builtin_amdgcn_wmma_f32_16x16x32_f16(false, a0, false, b0,
                                                   (short)0, acc00, false, false);
    acc10 = __builtin_amdgcn_wmma_f32_16x16x32_f16(false, a1, false, b0,
                                                   (short)0, acc10, false, false);
    acc01 = __builtin_amdgcn_wmma_f32_16x16x32_f16(false, a0, false, b1,
                                                   (short)0, acc01, false, false);
    acc11 = __builtin_amdgcn_wmma_f32_16x16x32_f16(false, a1, false, b1,
                                                   (short)0, acc11, false, false);

    // stores: D VGPR g -> channel cobase+half*8+g (8 consecutive), lane -> pixel
    const int cA = co0 + half * 8;
    const int cB = co0 + 16 + half * 8;
    h8 oA, oB;
    _Float16* p0 = out + (((size_t)img * H + y0) * W + xc0) * Cout;
    _Float16* p1 = out + (((size_t)img * H + y1) * W + xc1) * Cout;
#pragma unroll
    for (int g = 0; g < 8; ++g) {
        oA[g] = (_Float16)fmaxf(acc00[g] + bias[cA + g], 0.0f);
        oB[g] = (_Float16)fmaxf(acc10[g] + bias[cB + g], 0.0f);
    }
    *(h8*)(p0 + cA) = oA;
    *(h8*)(p0 + cB) = oB;
#pragma unroll
    for (int g = 0; g < 8; ++g) {
        oA[g] = (_Float16)fmaxf(acc01[g] + bias[cA + g], 0.0f);
        oB[g] = (_Float16)fmaxf(acc11[g] + bias[cB + g], 0.0f);
    }
    *(h8*)(p1 + cA) = oA;
    *(h8*)(p1 + cB) = oB;
}

// ------------------------- small helper kernels ----------------------------

// frames f32 NCHW [40][3][256][256] -> im2col fp16 [40*65536][32] (K padded 27->32)
__global__ void im2col_l1(const float* __restrict__ fr, _Float16* __restrict__ col,
                          long total)
{
    for (long idx = blockIdx.x * 256L + threadIdx.x; idx < total;
         idx += (long)gridDim.x * 256L) {
        const int j = (int)(idx & 31);
        const long pix = idx >> 5;
        const int x = (int)(pix & 255);
        const int y = (int)((pix >> 8) & 255);
        const int img = (int)(pix >> 16);
        float v = 0.0f;
        if (j < 27) {
            const int tap = j / 3, ci = j % 3;
            const int yy = y + tap / 3 - 1;
            const int xx = x + tap % 3 - 1;
            if (yy >= 0 && yy < 256 && xx >= 0 && xx < 256)
                v = fr[(((long)img * 3 + ci) * 256 + yy) * 256 + xx];
        }
        col[idx] = (_Float16)v;
    }
}

// 2x2 maxpool stride 2, NHWC fp16
__global__ void maxpool2(const _Float16* __restrict__ in, _Float16* __restrict__ out,
                         int H, int W, int C, long total)
{
    const int Ho = H >> 1, Wo = W >> 1;
    for (long idx = blockIdx.x * 256L + threadIdx.x; idx < total;
         idx += (long)gridDim.x * 256L) {
        long t = idx;
        const int c = (int)(t % C); t /= C;
        const int xo = (int)(t % Wo); t /= Wo;
        const int yo = (int)(t % Ho);
        const int img = (int)(t / Ho);
        const _Float16* p = in + (((size_t)img * H + 2 * yo) * W + 2 * xo) * C + c;
        float m = fmaxf(fmaxf((float)p[0], (float)p[C]),
                        fmaxf((float)p[(size_t)W * C], (float)p[(size_t)W * C + C]));
        out[idx] = (_Float16)m;
    }
}

// conv weight OIHW f32 -> [O][9*I] fp16 with k=(rs)*I+i
__global__ void prep_conv_w3(const float* __restrict__ w, _Float16* __restrict__ w16,
                             int I, long total)
{
    const long K = 9L * I;
    for (long idx = blockIdx.x * 256L + threadIdx.x; idx < total;
         idx += (long)gridDim.x * 256L) {
        const int rs = (int)(idx % 9);
        long t = idx / 9;
        const int i = (int)(t % I);
        const long m = t / I;
        w16[m * K + (long)rs * I + i] = (_Float16)w[idx];
    }
}

// layer-1 weights [64][3][3][3] -> [64][32] fp16 (zero padded K 27->32)
__global__ void prep_conv_w1(const float* __restrict__ w, _Float16* __restrict__ w16)
{
    const int idx = blockIdx.x * 256 + threadIdx.x;
    if (idx >= 64 * 32) return;
    const int m = idx >> 5, j = idx & 31;
    float v = 0.0f;
    if (j < 27) v = w[(m * 3 + (j % 3)) * 9 + (j / 3)];
    w16[idx] = (_Float16)v;
}

__global__ void prep_lin(const float* __restrict__ w, _Float16* __restrict__ w16,
                         long total)
{
    for (long i = blockIdx.x * 256L + threadIdx.x; i < total;
         i += (long)gridDim.x * 256L)
        w16[i] = (_Float16)w[i];
}

// d1_w [1024][8192]: remap columns NCHW-flatten (c*64+pix) -> NHWC (pix*128+c)
__global__ void prep_d1(const float* __restrict__ w, _Float16* __restrict__ w16,
                        long total)
{
    for (long idx = blockIdx.x * 256L + threadIdx.x; idx < total;
         idx += (long)gridDim.x * 256L) {
        const long m = idx >> 13;
        const int dst = (int)(idx & 8191);
        const int pix = dst >> 7, c = dst & 127;
        w16[idx] = (_Float16)w[(m << 13) + c * 64 + pix];
    }
}

// LayerNorm over D (+optional residual, relu, positional emb added after relu)
__global__ __launch_bounds__(256)
void ln_kernel(const float* __restrict__ x, const float* __restrict__ res,
               const float* __restrict__ w, const float* __restrict__ b,
               const float* __restrict__ pos, float* __restrict__ y32,
               _Float16* __restrict__ y16, int D, int Spos, int do_relu)
{
    const int row = blockIdx.x;
    const float* xr = x + (size_t)row * D;
    const float* rr = res ? res + (size_t)row * D : nullptr;
    __shared__ float s1[256], s2[256];
    float a = 0.0f, q = 0.0f;
    for (int i = threadIdx.x; i < D; i += 256) {
        float v = xr[i] + (rr ? rr[i] : 0.0f);
        a += v; q += v * v;
    }
    s1[threadIdx.x] = a; s2[threadIdx.x] = q;
    __syncthreads();
    for (int o = 128; o > 0; o >>= 1) {
        if (threadIdx.x < o) {
            s1[threadIdx.x] += s1[threadIdx.x + o];
            s2[threadIdx.x] += s2[threadIdx.x + o];
        }
        __syncthreads();
    }
    const float mean = s1[0] / D;
    const float var = s2[0] / D - mean * mean;
    const float rstd = rsqrtf(var + 1e-5f);
    const float* pr = pos ? pos + (size_t)(row % Spos) * D : nullptr;
    for (int i = threadIdx.x; i < D; i += 256) {
        float v = xr[i] + (rr ? rr[i] : 0.0f);
        float y = (v - mean) * rstd * w[i] + b[i];
        if (do_relu) y = fmaxf(y, 0.0f);
        if (pr) y += pr[i];
        y32[(size_t)row * D + i] = y;
        y16[(size_t)row * D + i] = (_Float16)y;
    }
}

// Attention (faithful to reference: attends over the batch axis, L=M=4).
// qkv f32 [40][3072]; block = (s,h); writes o [40][1024] f32+f16.
__global__ __launch_bounds__(128)
void attn_kernel(const float* __restrict__ qkv, float* __restrict__ o32,
                 _Float16* __restrict__ o16)
{
    const int s = blockIdx.x % 10;
    const int h = blockIdx.x / 10;
    const int qo = h * 128;
    __shared__ float sc[4][4];
    const int t = threadIdx.x;
    if (t < 16) {
        const int l = t >> 2, m = t & 3;
        const float* qp = qkv + (size_t)(l * 10 + s) * 3072 + qo;
        const float* kp = qkv + (size_t)(m * 10 + s) * 3072 + 1024 + qo;
        float sum = 0.0f;
        for (int d = 0; d < 128; ++d) sum += qp[d] * kp[d];
        sc[l][m] = sum * 0.08838834764831845f;   // 1/sqrt(128)
    }
    __syncthreads();
    if (t < 4) {
        float mx = sc[t][0];
        for (int m = 1; m < 4; ++m) mx = fmaxf(mx, sc[t][m]);
        float e[4], den = 0.0f;
        for (int m = 0; m < 4; ++m) { e[m] = expf(sc[t][m] - mx); den += e[m]; }
        for (int m = 0; m < 4; ++m) sc[t][m] = e[m] / den;
    }
    __syncthreads();
    const int d = t;
    for (int l = 0; l < 4; ++l) {
        float o = 0.0f;
        for (int m = 0; m < 4; ++m)
            o += sc[l][m] * qkv[(size_t)(m * 10 + s) * 3072 + 2048 + qo + d];
        const size_t oi = (size_t)(l * 10 + s) * 1024 + qo + d;
        o32[oi] = o;
        o16[oi] = (_Float16)o;
    }
}

__global__ void gelu_kernel(const float* __restrict__ x, float* __restrict__ y, int n)
{
    const int i = blockIdx.x * 256 + threadIdx.x;
    if (i < n) {
        float v = x[i];
        y[i] = 0.5f * v * (1.0f + erff(v * 0.7071067811865475f));
    }
}

// p2[n][m] = fp2_b[m] + sum_k16 g[n][k]*fp2_w[m][k]   (K=16 -> scalar)
__global__ void fp2_kernel(const float* __restrict__ g, const float* __restrict__ w,
                           const float* __restrict__ b, float* __restrict__ p2)
{
    const int idx = blockIdx.x * 256 + threadIdx.x;
    if (idx >= 40 * 1024) return;
    const int n = idx >> 10, m = idx & 1023;
    float s = b[m];
    for (int k = 0; k < 16; ++k) s += g[n * 16 + k] * w[m * 16 + k];
    p2[idx] = s;
}

// final head: dp3 (64->4) + sigmoid -> d_out f32 [4,10,4]
__global__ void head_kernel(const float* __restrict__ d64, const float* __restrict__ w,
                            const float* __restrict__ b, float* __restrict__ out)
{
    const int idx = blockIdx.x * 256 + threadIdx.x;
    if (idx >= 160) return;
    const int n = idx >> 2, c = idx & 3;
    float s = b[c];
    for (int k = 0; k < 64; ++k) s += d64[n * 64 + k] * w[c * 64 + k];
    out[idx] = 1.0f / (1.0f + expf(-s));
}

// ---------------------------------------------------------------------------
extern "C" void kernel_launch(void* const* d_in, const int* in_sizes, int n_in,
                              void* d_out, int out_size, void* d_ws, size_t ws_size,
                              hipStream_t stream)
{
    (void)in_sizes; (void)n_in; (void)out_size; (void)ws_size;

    const float* frames = (const float*)d_in[0];
    const float* vw[13]; const float* vb[13];
    for (int i = 0; i < 13; ++i) { vw[i] = (const float*)d_in[1 + i];
                                   vb[i] = (const float*)d_in[14 + i]; }
    const float* cr_w = (const float*)d_in[27];  const float* cr_b = (const float*)d_in[28];
    const float* d1_w = (const float*)d_in[29];  const float* d1_b = (const float*)d_in[30];
    const float* n1_w = (const float*)d_in[31];  const float* n1_b = (const float*)d_in[32];
    const float* pos  = (const float*)d_in[33];
    const float* ipw  = (const float*)d_in[34];  const float* ipb  = (const float*)d_in[35];
    const float* opw  = (const float*)d_in[36];  const float* opb  = (const float*)d_in[37];
    const float* ln1w = (const float*)d_in[38];  const float* ln1b = (const float*)d_in[39];
    const float* fp1w = (const float*)d_in[40];  const float* fp1b = (const float*)d_in[41];
    const float* fp2w = (const float*)d_in[42];  const float* fp2b = (const float*)d_in[43];
    const float* ln2w = (const float*)d_in[44];  const float* ln2b = (const float*)d_in[45];
    const float* dp2w = (const float*)d_in[46];  const float* dp2b = (const float*)d_in[47];
    const float* dp3w = (const float*)d_in[48];  const float* dp3b = (const float*)d_in[49];

    static const int cin[13]  = {3, 64, 64, 128, 128, 256, 256, 256, 512, 512, 512, 512, 512};
    static const int csh[13]  = {0, 6, 6, 7, 7, 8, 8, 8, 9, 9, 9, 9, 9};
    static const int cout[13] = {64, 64, 128, 128, 256, 256, 256, 512, 512, 512, 512, 512, 512};
    static const int chw[13]  = {256, 256, 128, 128, 64, 64, 64, 32, 32, 32, 16, 16, 16};

    size_t off = 0;
    auto alloc = [&](size_t bytes) -> void* {
        off = (off + 255) & ~(size_t)255;
        void* p = (void*)((char*)d_ws + off);
        off += bytes;
        return p;
    };

    _Float16* wc16[13];
    for (int l = 0; l < 13; ++l) {
        const size_t K = (l == 0) ? 32 : (size_t)9 * cin[l];
        wc16[l] = (_Float16*)alloc((size_t)cout[l] * K * 2);
    }
    _Float16* crw16  = (_Float16*)alloc((size_t)128 * 512 * 2);
    _Float16* d1w16  = (_Float16*)alloc((size_t)1024 * 8192 * 2);
    _Float16* ipw16  = (_Float16*)alloc((size_t)3072 * 1024 * 2);
    _Float16* opw16  = (_Float16*)alloc((size_t)1024 * 1024 * 2);
    _Float16* fp1w16 = (_Float16*)alloc((size_t)16 * 1024 * 2);
    _Float16* dp2w16 = (_Float16*)alloc((size_t)64 * 1024 * 2);
    _Float16* bufA   = (_Float16*)alloc((size_t)40 * 256 * 256 * 64 * 2);
    _Float16* bufB   = (_Float16*)alloc((size_t)40 * 256 * 256 * 64 * 2);
    float*    x32    = (float*)alloc(40 * 1024 * 4);
    float*    xln32  = (float*)alloc(40 * 1024 * 4);
    _Float16* x16    = (_Float16*)alloc(40 * 1024 * 2);
    float*    qkv32  = (float*)alloc(40 * 3072 * 4);
    float*    att32  = (float*)alloc(40 * 1024 * 4);
    _Float16* att16  = (_Float16*)alloc(40 * 1024 * 2);
    float*    ao32   = (float*)alloc(40 * 1024 * 4);
    float*    y32    = (float*)alloc(40 * 1024 * 4);
    _Float16* y16    = (_Float16*)alloc(40 * 1024 * 2);
    float*    p1     = (float*)alloc(40 * 16 * 4);
    float*    g1     = (float*)alloc(40 * 16 * 4);
    float*    p2     = (float*)alloc(40 * 1024 * 4);
    float*    z32    = (float*)alloc(40 * 1024 * 4);
    _Float16* z16    = (_Float16*)alloc(40 * 1024 * 2);
    float*    d64    = (float*)alloc(40 * 64 * 4);

    // ---- weight preparation -------------------------------------------------
    prep_conv_w1<<<(64 * 32 + 255) / 256, 256, 0, stream>>>(vw[0], wc16[0]);
    for (int l = 1; l < 13; ++l) {
        const long total = (long)cout[l] * cin[l] * 9;
        prep_conv_w3<<<(int)((total + 255) / 256), 256, 0, stream>>>(vw[l], wc16[l],
                                                                     cin[l], total);
    }
    prep_lin<<<(128 * 512 + 255) / 256, 256, 0, stream>>>(cr_w, crw16, 128L * 512);
    prep_d1<<<(int)((1024L * 8192 + 255) / 256), 256, 0, stream>>>(d1_w, d1w16,
                                                                   1024L * 8192);
    prep_lin<<<(int)((3072L * 1024 + 255) / 256), 256, 0, stream>>>(ipw, ipw16, 3072L * 1024);
    prep_lin<<<(int)((1024L * 1024 + 255) / 256), 256, 0, stream>>>(opw, opw16, 1024L * 1024);
    prep_lin<<<(16 * 1024 + 255) / 256, 256, 0, stream>>>(fp1w, fp1w16, 16L * 1024);
    prep_lin<<<(64 * 1024 + 255) / 256, 256, 0, stream>>>(dp2w, dp2w16, 64L * 1024);

    // layer-1 im2col into bufB (overwritten by L1 conv output afterwards)
    {
        const long total = 40L * 65536 * 32;
        im2col_l1<<<(int)((total + 255) / 256), 256, 0, stream>>>(frames, bufB, total);
    }

    auto gemm = [&](const _Float16* A, const _Float16* B, const float* bias,
                    _Float16* o16, float* o32, int M, int N, int K, int relu) {
        dim3 g((N + 15) / 16, M / 16);
        gemm_tn_wmma<<<g, 32, 0, stream>>>(A, B, bias, o16, o32, M, N, K, relu);
    };
    auto conv = [&](const _Float16* in, int l, _Float16* out) {
        const int H = chw[l];
        dim3 g(40 * ((H * H) >> 5), cout[l] / 32);
        conv3x3_wmma<<<g, 32, 0, stream>>>(in, wc16[l], vb[l], out, H, H, csh[l], cout[l]);
    };
    auto pool = [&](const _Float16* in, _Float16* out, int H, int C) {
        const long total = 40L * (H / 2) * (H / 2) * C;
        maxpool2<<<(int)((total + 255) / 256), 256, 0, stream>>>(in, out, H, H, C, total);
    };

    // ---- VGG stack ----------------------------------------------------------
    gemm(wc16[0], bufB, vb[0], bufA, nullptr, 64, 40 * 65536, 32, 1); // conv1 via GEMM
    conv(bufA, 1, bufB);
    pool(bufB, bufA, 256, 64);
    conv(bufA, 2, bufB);
    conv(bufB, 3, bufA);
    pool(bufA, bufB, 128, 128);
    conv(bufB, 4, bufA);
    conv(bufA, 5, bufB);
    conv(bufB, 6, bufA);
    pool(bufA, bufB, 64, 256);
    conv(bufB, 7, bufA);
    conv(bufA, 8, bufB);
    conv(bufB, 9, bufA);
    pool(bufA, bufB, 32, 512);
    conv(bufB, 10, bufA);
    conv(bufA, 11, bufB);
    conv(bufB, 12, bufA);
    pool(bufA, bufB, 16, 512);                       // -> [40][8][8][512] in bufB

    // ---- classifier head (cr 1x1 conv + dense) ------------------------------
    gemm(crw16, bufB, cr_b, bufA, nullptr, 128, 2560, 512, 1);   // relu(cr) -> [40][8192]
    gemm(d1w16, bufA, d1_b, nullptr, x32, 1024, 40, 8192, 0);

    // LN -> relu -> +pos_emb
    ln_kernel<<<40, 256, 0, stream>>>(x32, nullptr, n1_w, n1_b, pos, xln32, x16, 1024, 10, 1);

    // ---- transformer block --------------------------------------------------
    gemm(ipw16, x16, ipb, nullptr, qkv32, 3072, 40, 1024, 0);
    attn_kernel<<<80, 128, 0, stream>>>(qkv32, att32, att16);
    gemm(opw16, att16, opb, nullptr, ao32, 1024, 40, 1024, 0);
    ln_kernel<<<40, 256, 0, stream>>>(xln32, ao32, ln1w, ln1b, nullptr, y32, y16, 1024, 10, 0);
    gemm(fp1w16, y16, fp1b, nullptr, p1, 16, 40, 1024, 0);
    gelu_kernel<<<(640 + 255) / 256, 256, 0, stream>>>(p1, g1, 640);
    fp2_kernel<<<(40 * 1024 + 255) / 256, 256, 0, stream>>>(g1, fp2w, fp2b, p2);
    ln_kernel<<<40, 256, 0, stream>>>(p2, nullptr, ln2w, ln2b, nullptr, z32, z16, 1024, 10, 0);
    gemm(dp2w16, z16, dp2b, nullptr, d64, 64, 40, 1024, 0);
    head_kernel<<<1, 256, 0, stream>>>(d64, dp3w, dp3b, (float*)d_out);
}
